// CosmicMultiHeadAttention_70600672411790
// MI455X (gfx1250) — matrix operand order repaired and checked by reading, not compiled, not run on hardware
//
#include <hip/hip_runtime.h>
#include <hip/hip_bf16.h>
#include <cstdint>

#define HQ_  128
#define HKV_ 16
#define D_   128
#define BS_  128
#define B_   2
#define S_   2048
#define E_   8192
#define NB_  (S_/BS_)   // 16

typedef __attribute__((ext_vector_type(16))) __bf16 v16bf;
typedef __attribute__((ext_vector_type(8)))  float  v8f;

union FragBF {
  v16bf v;
  uint4 q[2];
  unsigned short u[16];
};

__device__ __forceinline__ unsigned short f2bf(float x) {
  union { float f; unsigned int u; } c; c.f = x;
  unsigned int u = c.u;
  unsigned int r = u + 0x7fffu + ((u >> 16) & 1u);   // round-to-nearest-even
  return (unsigned short)(r >> 16);
}
__device__ __forceinline__ float bf2f(unsigned short h) {
  union { unsigned int u; float f; } c; c.u = ((unsigned int)h) << 16;
  return c.f;
}

// ---------------------------------------------------------------------------
// C[M x N] = A[M x K] @ W[K x N] + bias.  A is fp32 or bf16(ushort); W fp32.
// 128x128 tile per workgroup, 8 waves, each wave = 16x128 strip (8 WMMA tiles).
// M is blockIdx.x (fastest) so concurrently-running workgroups share the same
// W column strip through the 192MB L2; x/A stays L2-resident.
// Software pipeline: global loads for chunk k+1 issued into registers before
// the WMMA block for chunk k (latency hidden behind 8 WMMAs per wave), staged into the
// single LDS buffer after the post-compute barrier.
// ---------------------------------------------------------------------------
template <typename AT>
__global__ __launch_bounds__(256)
void gemm_bf16_wmma(const AT* __restrict__ A, int lda,
                    const float* __restrict__ W, int ldw,
                    const float* __restrict__ bias,
                    float* __restrict__ Cf, int ldcf,
                    unsigned short* __restrict__ Cb, int ldcb,
                    int K)
{
  __shared__ unsigned short Abuf[128 * 40];
  __shared__ unsigned short Bbuf[128 * 40];

  const int tid  = threadIdx.x;
  const int lane = tid & 31;
  const int wave = tid >> 5;
  const int half = lane >> 4;
  const int c16  = lane & 15;
  const int m0 = blockIdx.x * 128;   // M fastest -> W-tile reuse in L2
  const int n0 = blockIdx.y * 128;

  v8f acc[8] = {};

  float4 aReg[4];   // fp32-A staging
  uint4  aRegH[2];  // bf16-A staging
  float4 wReg[4];

  auto loadA = [&](int k0) {
    if constexpr (sizeof(AT) == 4) {
      const float* Af = (const float*)A;
      #pragma unroll
      for (int i = 0; i < 4; ++i) {
        int f = tid + i * 256;             // 1024 float4 chunks (128x32)
        int row = f >> 3, c4 = f & 7;
        aReg[i] = *(const float4*)(Af + (size_t)(m0 + row) * lda + k0 + c4 * 4);
      }
    } else {
      const unsigned short* Ab = (const unsigned short*)A;
      #pragma unroll
      for (int i = 0; i < 2; ++i) {
        int f = tid + i * 256;             // 512 chunks of 8 bf16
        int row = f >> 2, c8 = f & 3;
        aRegH[i] = *(const uint4*)(Ab + (size_t)(m0 + row) * lda + k0 + c8 * 8);
      }
    }
  };
  auto loadW = [&](int k0) {
    #pragma unroll
    for (int i = 0; i < 4; ++i) {
      int f = tid + i * 256;               // 1024 float4 chunks (32x128)
      int kk = f >> 5, n4 = f & 31;
      wReg[i] = *(const float4*)(W + (size_t)(k0 + kk) * ldw + n0 + n4 * 4);
    }
  };
  auto stage = [&]() {
    if constexpr (sizeof(AT) == 4) {
      #pragma unroll
      for (int i = 0; i < 4; ++i) {
        int f = tid + i * 256;
        int row = f >> 3, c4 = f & 7;
        unsigned short* dst = &Abuf[row * 40 + c4 * 4];
        dst[0] = f2bf(aReg[i].x); dst[1] = f2bf(aReg[i].y);
        dst[2] = f2bf(aReg[i].z); dst[3] = f2bf(aReg[i].w);
      }
    } else {
      #pragma unroll
      for (int i = 0; i < 2; ++i) {
        int f = tid + i * 256;
        int row = f >> 2, c8 = f & 3;
        *(uint4*)&Abuf[row * 40 + c8 * 8] = aRegH[i];
      }
    }
    #pragma unroll
    for (int i = 0; i < 4; ++i) {
      int f = tid + i * 256;
      int kk = f >> 5, n4 = f & 31;
      Bbuf[(n4 * 4 + 0) * 40 + kk] = f2bf(wReg[i].x);
      Bbuf[(n4 * 4 + 1) * 40 + kk] = f2bf(wReg[i].y);
      Bbuf[(n4 * 4 + 2) * 40 + kk] = f2bf(wReg[i].z);
      Bbuf[(n4 * 4 + 3) * 40 + kk] = f2bf(wReg[i].w);
    }
  };

  loadA(0); loadW(0);
  stage();
  __syncthreads();

  for (int k0 = 0; k0 < K; k0 += 32) {
    const bool more = (k0 + 32) < K;
    if (more) { loadA(k0 + 32); loadW(k0 + 32); }  // in flight during WMMAs

    // A frag per ISA layout (lane<16: K 0-7,16-23; lane>=16: K 8-15,24-31)
    FragBF a;
    const int arow = wave * 16 + c16;
    a.q[0] = *(const uint4*)&Abuf[arow * 40 + half * 8];
    a.q[1] = *(const uint4*)&Abuf[arow * 40 + 16 + half * 8];
    #pragma unroll
    for (int t = 0; t < 8; ++t) {
      FragBF b;
      const int bcol = t * 16 + c16;   // B frag: lane=N, K = half*8..+7, 16+half*8..+7
      b.q[0] = *(const uint4*)&Bbuf[bcol * 40 + half * 8];
      b.q[1] = *(const uint4*)&Bbuf[bcol * 40 + 16 + half * 8];
      acc[t] = __builtin_amdgcn_wmma_f32_16x16x32_bf16(
          false, a.v, false, b.v, (short)0, acc[t], false, false);
    }
    __syncthreads();                 // all LDS reads of this chunk done
    if (more) stage();               // overwrite LDS with next chunk
    __syncthreads();
  }

  // epilogue: C layout = lane holds rows {0..7}+8*half, col = lane&15
  const int mrow = m0 + wave * 16 + half * 8;
  #pragma unroll
  for (int t = 0; t < 8; ++t) {
    const int cg = n0 + t * 16 + c16;
    const float bb = bias ? bias[cg] : 0.0f;
    #pragma unroll
    for (int r = 0; r < 8; ++r) {
      float v = acc[t][r] + bb;
      if (Cf) Cf[(size_t)(mrow + r) * ldcf + cg] = v;
      if (Cb) Cb[(size_t)(mrow + r) * ldcb + cg] = f2bf(v);
    }
  }
}

// ---------------------------------------------------------------------------
// RoPE, in-place on bf16 [B,S,H,D]; optionally emit fp32 copy (for k output).
// One thread per (b,s,h,i) with i in [0,64): handles the (i, i+64) pair.
// ---------------------------------------------------------------------------
__global__ __launch_bounds__(256)
void rope_kernel(unsigned short* __restrict__ data,
                 float* __restrict__ outf,
                 int H, int total)
{
  int idx = blockIdx.x * 256 + threadIdx.x;
  if (idx >= total) return;
  int i = idx & 63;
  int t = idx >> 6;
  int h = t % H;
  int s = (t / H) % S_;
  int b = t / (H * S_);
  size_t base = ((size_t)(b * S_ + s) * H + h) * D_;
  float x1 = bf2f(data[base + i]);
  float x2 = bf2f(data[base + 64 + i]);
  const float kLogBaseOver = 9.210340371976184f / 64.0f;  // ln(10000)/64
  float invf = __expf(-(float)i * kLogBaseOver);
  float ang = (float)s * invf;
  float c = cosf(ang), sn = sinf(ang);
  float o1 = x1 * c - x2 * sn;
  float o2 = x2 * c + x1 * sn;
  data[base + i]      = f2bf(o1);
  data[base + 64 + i] = f2bf(o2);
  if (outf) { outf[base + i] = o1; outf[base + 64 + i] = o2; }
}

// ---------------------------------------------------------------------------
// Block-diagonal attention: one workgroup per (b, block, q-head).
// Phase 1: S = Q Kᵀ (K stored [n][d] == B layout), softmax via shfl.
// Phase 2: P staged to LDS (C->A layout transpose), V staged transposed
//          [d][key] into LDS, O = P V, normalize, write bf16 attn_out.
// ---------------------------------------------------------------------------
__global__ __launch_bounds__(256)
void attn_kernel(const unsigned short* __restrict__ q,
                 const unsigned short* __restrict__ k,
                 const unsigned short* __restrict__ v,
                 unsigned short* __restrict__ o)
{
  __shared__ unsigned short smA[128 * 136];  // Q, then P
  __shared__ unsigned short smB[128 * 136];  // K, then V^T

  const int tid  = threadIdx.x;
  const int lane = tid & 31;
  const int wave = tid >> 5;
  const int half = lane >> 4;
  const int c16  = lane & 15;

  const int id  = blockIdx.x;
  const int h   = id & (HQ_ - 1);
  const int blk = (id >> 7) & (NB_ - 1);
  const int b   = id >> 11;
  const int kvh = h >> 3;                    // G = HQ/HKV = 8

  const size_t qbase = ((size_t)(b * S_ + blk * BS_) * HQ_  + h)   * D_;
  const size_t kbase = ((size_t)(b * S_ + blk * BS_) * HKV_ + kvh) * D_;

  // stage Q [m][d] and K [n][d]
  #pragma unroll
  for (int i = 0; i < 8; ++i) {
    int f = tid + i * 256;                   // 2048 chunks of 8 bf16
    int row = f >> 4;
    int c8  = f & 15;
    uint4 qa = *(const uint4*)(q + qbase + (size_t)row * (HQ_ * D_) + c8 * 8);
    *(uint4*)&smA[row * 136 + c8 * 8] = qa;
    uint4 ka = *(const uint4*)(k + kbase + (size_t)row * (HKV_ * D_) + c8 * 8);
    *(uint4*)&smB[row * 136 + c8 * 8] = ka;
  }
  __syncthreads();

  // ---- S = Q K^T ----
  v8f acc[8] = {};
  const int arow = wave * 16 + c16;
  #pragma unroll
  for (int kc = 0; kc < 4; ++kc) {
    FragBF a;
    a.q[0] = *(const uint4*)&smA[arow * 136 + kc * 32 + half * 8];
    a.q[1] = *(const uint4*)&smA[arow * 136 + kc * 32 + 16 + half * 8];
    #pragma unroll
    for (int t = 0; t < 8; ++t) {
      FragBF bf;
      int bcol = t * 16 + c16;
      bf.q[0] = *(const uint4*)&smB[bcol * 136 + kc * 32 + half * 8];
      bf.q[1] = *(const uint4*)&smB[bcol * 136 + kc * 32 + 16 + half * 8];
      acc[t] = __builtin_amdgcn_wmma_f32_16x16x32_bf16(
          false, a.v, false, bf.v, (short)0, acc[t], false, false);
    }
  }

  // ---- softmax over 128 keys; lane holds rows {0..7}+8*half, cols via lanes ----
  const float scale = 0.08838834764831845f;  // 1/sqrt(128)
  float mx[8], sm[8];
  #pragma unroll
  for (int r = 0; r < 8; ++r) {
    float m = -3.0e38f;
    #pragma unroll
    for (int t = 0; t < 8; ++t) m = fmaxf(m, acc[t][r]);
    #pragma unroll
    for (int d = 1; d < 16; d <<= 1) m = fmaxf(m, __shfl_xor(m, d));
    mx[r] = m * scale;
  }
  #pragma unroll
  for (int t = 0; t < 8; ++t)
    #pragma unroll
    for (int r = 0; r < 8; ++r)
      acc[t][r] = __expf(acc[t][r] * scale - mx[r]);
  #pragma unroll
  for (int r = 0; r < 8; ++r) {
    float s = 0.0f;
    #pragma unroll
    for (int t = 0; t < 8; ++t) s += acc[t][r];
    #pragma unroll
    for (int d = 1; d < 16; d <<= 1) s += __shfl_xor(s, d);
    sm[r] = s;
  }
  __syncthreads();  // everyone done reading Q/K from LDS

  // ---- stage P (exp, bf16) into smA [m][key]; stage V^T into smB [d][key] ----
  {
    const int prow = wave * 16 + half * 8;
    #pragma unroll
    for (int t = 0; t < 8; ++t)
      #pragma unroll
      for (int r = 0; r < 8; ++r)
        smA[(prow + r) * 136 + t * 16 + c16] = f2bf(acc[t][r]);
  }
  #pragma unroll
  for (int i = 0; i < 8; ++i) {
    int f = tid + i * 256;
    int key = f >> 4;
    int c8  = f & 15;
    union { uint4 qv; unsigned short u[8]; } tmp;
    tmp.qv = *(const uint4*)(v + kbase + (size_t)key * (HKV_ * D_) + c8 * 8);
    #pragma unroll
    for (int j = 0; j < 8; ++j)
      smB[(c8 * 8 + j) * 136 + key] = tmp.u[j];
  }
  __syncthreads();

  // ---- O = P V ----
  v8f acc2[8] = {};
  #pragma unroll
  for (int kc = 0; kc < 4; ++kc) {
    FragBF a;
    a.q[0] = *(const uint4*)&smA[arow * 136 + kc * 32 + half * 8];
    a.q[1] = *(const uint4*)&smA[arow * 136 + kc * 32 + 16 + half * 8];
    #pragma unroll
    for (int t = 0; t < 8; ++t) {
      FragBF bf;
      int bcol = t * 16 + c16;            // output d-index group
      bf.q[0] = *(const uint4*)&smB[bcol * 136 + kc * 32 + half * 8];
      bf.q[1] = *(const uint4*)&smB[bcol * 136 + kc * 32 + 16 + half * 8];
      acc2[t] = __builtin_amdgcn_wmma_f32_16x16x32_bf16(
          false, a.v, false, bf.v, (short)0, acc2[t], false, false);
    }
  }

  // ---- normalize + write attn_out bf16 (same layout as q) ----
  const int orow0 = wave * 16 + half * 8;
  #pragma unroll
  for (int t = 0; t < 8; ++t) {
    int dcol = t * 16 + c16;
    #pragma unroll
    for (int r = 0; r < 8; ++r) {
      float val = acc2[t][r] / sm[r];
      o[qbase + (size_t)(orow0 + r) * (HQ_ * D_) + dcol] = f2bf(val);
    }
  }
}

// ---------------------------------------------------------------------------
extern "C" void kernel_launch(void* const* d_in, const int* in_sizes, int n_in,
                              void* d_out, int out_size, void* d_ws, size_t ws_size,
                              hipStream_t stream)
{
  const float* x  = (const float*)d_in[0];
  const float* Wq = (const float*)d_in[1];
  const float* bq = (const float*)d_in[2];
  const float* Wk = (const float*)d_in[3];
  const float* bk = (const float*)d_in[4];
  const float* Wv = (const float*)d_in[5];
  const float* bv = (const float*)d_in[6];
  const float* Wo = (const float*)d_in[7];
  const float* bo = (const float*)d_in[8];

  float* out  = (float*)d_out;                          // (B,S,E)
  float* kout = out  + (size_t)B_ * S_ * E_;            // (B,S,HKV,D)
  float* vout = kout + (size_t)B_ * S_ * HKV_ * D_;     // (B,S,HKV,D)

  unsigned short* ws_q = (unsigned short*)d_ws;         // [B*S][HQ*D] bf16; reused as attn_out
  unsigned short* ws_k = ws_q + (size_t)B_ * S_ * HQ_  * D_;
  unsigned short* ws_v = ws_k + (size_t)B_ * S_ * HKV_ * D_;

  const int M = B_ * S_;  // 4096

  // q = x@Wq + bq  -> bf16 ws      (grid: M fastest for W reuse in L2)
  gemm_bf16_wmma<float><<<dim3(M / 128, (HQ_ * D_) / 128), 256, 0, stream>>>(
      x, E_, Wq, HQ_ * D_, bq, nullptr, 0, ws_q, HQ_ * D_, E_);
  // k = x@Wk + bk  -> bf16 ws (pre-rope)
  gemm_bf16_wmma<float><<<dim3(M / 128, (HKV_ * D_) / 128), 256, 0, stream>>>(
      x, E_, Wk, HKV_ * D_, bk, nullptr, 0, ws_k, HKV_ * D_, E_);
  // v = x@Wv + bv  -> bf16 ws AND fp32 output
  gemm_bf16_wmma<float><<<dim3(M / 128, (HKV_ * D_) / 128), 256, 0, stream>>>(
      x, E_, Wv, HKV_ * D_, bv, vout, HKV_ * D_, ws_v, HKV_ * D_, E_);

  // RoPE (q in place; k in place + fp32 k output)
  {
    int totq = B_ * S_ * HQ_ * 64;
    rope_kernel<<<totq / 256, 256, 0, stream>>>(ws_q, nullptr, HQ_, totq);
    int totk = B_ * S_ * HKV_ * 64;
    rope_kernel<<<totk / 256, 256, 0, stream>>>(ws_k, kout, HKV_, totk);
  }

  // block-diagonal attention; attn_out overwrites ws_q in place (safe: each
  // workgroup stages its own q slice to LDS before writing the same slice)
  attn_kernel<<<B_ * NB_ * HQ_, 256, 0, stream>>>(ws_q, ws_k, ws_v, ws_q);

  // out = attn_out @ Wo + bo
  gemm_bf16_wmma<unsigned short><<<dim3(M / 128, E_ / 128), 256, 0, stream>>>(
      ws_q, HQ_ * D_, Wo, E_, bo, out, E_, nullptr, 0, HQ_ * D_);
}
